// AttentionGRUDecoder_1554778161654
// MI455X (gfx1250) — matrix operand order repaired and checked by reading, not compiled, not run on hardware
//
#include <hip/hip_runtime.h>
#include <hip/hip_bf16.h>
#include <math.h>

typedef __attribute__((ext_vector_type(16))) __bf16 v16bf;
typedef __attribute__((ext_vector_type(8)))  __bf16 v8bf;
typedef __attribute__((ext_vector_type(8)))  float  v8f;

#define B_  256
#define S_  2048
#define H_  256
#define TS  128          // s-rows per block
#define LDE 264          // padded LDS row stride (elements): breaks bank alignment

// Branchless tanh: single V_TANH_F32 (TRANS, co-executes with XDL WMMA) when the
// toolchain exposes it; otherwise ~5 branchless VALU/TRANS ops.
__device__ __forceinline__ float fast_tanh(float x) {
#if __has_builtin(__builtin_amdgcn_tanhf)
    return __builtin_amdgcn_tanhf(x);
#else
#if __has_builtin(__builtin_amdgcn_exp2f)
    const float e = __builtin_amdgcn_exp2f(x * 2.8853900817779268f); // exp(2x)
#else
    const float e = __expf(2.0f * x);
#endif
#if __has_builtin(__builtin_amdgcn_rcpf)
    return 1.0f - 2.0f * __builtin_amdgcn_rcpf(e + 1.0f);
#else
    return 1.0f - 2.0f / (e + 1.0f);
#endif
#endif
}

// fp32x4 -> bf16x4 packed 8-byte LDS store helper
__device__ __forceinline__ void store_bf4(__bf16* d, float4 v) {
    union { __bf16 h[4]; uint2 u; } p;
    p.h[0] = (__bf16)v.x; p.h[1] = (__bf16)v.y;
    p.h[2] = (__bf16)v.z; p.h[3] = (__bf16)v.w;
    *(uint2*)d = p.u;     // d is 8-byte aligned (col multiple of 4, stride 16B-aligned)
}

// ---------------------------------------------------------------------------
// Kernel 1: qb[b,h] = hidden[0,b,:] . Wq[h,:] + attn_b[h]   (tiny GEMV batch)
// ---------------------------------------------------------------------------
__global__ __launch_bounds__(256) void qb_kernel(
    const float* __restrict__ hid, const float* __restrict__ Wq,
    const float* __restrict__ attn_b, float* __restrict__ qb)
{
    __shared__ float sh[H_];
    const int b = blockIdx.x, t = threadIdx.x;
    sh[t] = hid[(size_t)b * H_ + t];
    __syncthreads();
    const float* wr = Wq + (size_t)t * H_;
    float s = 0.f;
#pragma unroll 8
    for (int e = 0; e < H_; ++e) s += wr[e] * sh[e];
    qb[(size_t)b * H_ + t] = s + attn_b[t];
}

// ---------------------------------------------------------------------------
// Kernel 2 (the big one): fused  k = enc@Wk^T  ->  score = Ws.tanh(k+q+b)
//   -> tile-local softmax -> partial context, with enc_out read ONCE.
// Grid: (S/TS, B). Block: 256 threads = 8 waves (wave32).
// ---------------------------------------------------------------------------
__global__ __launch_bounds__(256) void score_ctx_kernel(
    const float* __restrict__ enc, const float* __restrict__ Wk,
    const float* __restrict__ qb,  const float* __restrict__ Ws,
    float* __restrict__ cpart, float* __restrict__ mv, float* __restrict__ lv)
{
    extern __shared__ __align__(16) char smem[];
    __bf16* encs = (__bf16*)smem;              // TS  x LDE bf16
    __bf16* wks  = encs + TS * LDE;            // H_  x LDE bf16
    float*  sc   = (float*)(wks + H_ * LDE);   // TS scores
    float*  red  = sc + TS;                    // 256 reduction scratch

    const int t    = threadIdx.x;
    const int b    = blockIdx.y;
    const int tile = blockIdx.x;
    const int s0   = tile * TS;
    const float* encb = enc + ((size_t)b * S_ + s0) * H_;

    // --- stage enc tile fp32 -> bf16 LDS (coalesced b128 global, b64 LDS stores) ---
    for (int i = t; i < TS * (H_ / 4); i += 256) {
        const int row = i >> 6, c4 = (i & 63) << 2;
        store_bf4(encs + row * LDE + c4, *(const float4*)(encb + (size_t)row * H_ + c4));
    }
    // --- stage full Wk fp32 -> bf16 LDS (shared by all 8 waves, all n-tiles) ---
    for (int i = t; i < H_ * (H_ / 4); i += 256) {
        const int row = i >> 6, c4 = (i & 63) << 2;
        store_bf4(wks + row * LDE + c4, *(const float4*)(Wk + (size_t)row * H_ + c4));
    }
    __syncthreads();

    const int wave = t >> 5, lane = t & 31;
    const int half = lane >> 4, lrow = lane & 15;
    const int sw   = wave * 16;                // this wave's 16 s-rows

    union F16 { v16bf v; v8bf h[2]; };

    // --- preload this wave's 8 A fragments (16x32 bf16 each) into registers ---
    // ISA 7.12.2: lane<16 -> K{0..7,16..23}, lane>=16 -> K{8..15,24..31}, row M=lrow.
    F16 Af[8];
    {
        const __bf16* arow = encs + (sw + lrow) * LDE + half * 8;
#pragma unroll
        for (int kt = 0; kt < 8; ++kt) {
            Af[kt].h[0] = *(const v8bf*)(arow + kt * 32);
            Af[kt].h[1] = *(const v8bf*)(arow + kt * 32 + 16);
        }
    }
    // --- preload per-lane q+attn_b and Ws for all 16 n-tiles (no loads in loop) ---
    const float* qbb = qb + (size_t)b * H_;
    float qvs[16], wvs[16];
#pragma unroll
    for (int i = 0; i < 16; ++i) { qvs[i] = qbb[i * 16 + lrow]; wvs[i] = Ws[i * 16 + lrow]; }

    float acc[8];
#pragma unroll
    for (int i = 0; i < 8; ++i) acc[i] = 0.f;

    for (int n0i = 0; n0i < 16; ++n0i) {       // 16 N-tiles of kmat columns
        // B fragment: 32x16 bf16, B[e,h]=Wk[h,e]: lane n = column h,
        // lanes<16 hold K {0..7,16..23}, lanes>=16 hold K {8..15,24..31}.
        // Issue ALL 16 ds_loads first -> one dscnt wait -> 8 back-to-back WMMAs.
        const __bf16* brow = wks + (n0i * 16 + lrow) * LDE + half * 8;
        F16 Bf[8];
#pragma unroll
        for (int kt = 0; kt < 8; ++kt) {
            Bf[kt].h[0] = *(const v8bf*)(brow + kt * 32);
            Bf[kt].h[1] = *(const v8bf*)(brow + kt * 32 + 16);
        }
        v8f c = {};
#pragma unroll
        for (int kt = 0; kt < 8; ++kt)
            c = __builtin_amdgcn_wmma_f32_16x16x32_bf16(
                    false, Af[kt].v, false, Bf[kt].v, (short)0, c, false, false);
        // score contribution: tanh(k + q + attn_b) * Ws  (qb already has attn_b;
        // bs is softmax-shift-invariant and dropped). N index per lane = lrow.
        const float qv = qvs[n0i], wv = wvs[n0i];
#pragma unroll
        for (int i = 0; i < 8; ++i) acc[i] += fast_tanh(c[i] + qv) * wv;
    }

    // reduce over N within each 16-lane half (D layout: vgpr i -> M=i / M=8+i)
#pragma unroll
    for (int i = 0; i < 8; ++i) {
        float v = acc[i];
        v += __shfl_xor(v, 1, 32);
        v += __shfl_xor(v, 2, 32);
        v += __shfl_xor(v, 4, 32);
        v += __shfl_xor(v, 8, 32);
        acc[i] = v;
    }
    if (lrow == 0) {
#pragma unroll
        for (int i = 0; i < 8; ++i) sc[sw + half * 8 + i] = acc[i];
    }
    __syncthreads();

    // --- tile-local softmax (max, exp, sum) ---
    red[t] = (t < TS) ? sc[t] : -3.402823466e38f;
    __syncthreads();
    for (int off = 128; off > 0; off >>= 1) {
        if (t < off) red[t] = fmaxf(red[t], red[t + off]);
        __syncthreads();
    }
    const float mt = red[0];
    __syncthreads();
    const float e = (t < TS) ? __expf(sc[t] - mt) : 0.f;
    if (t < TS) sc[t] = e;
    red[t] = e;
    __syncthreads();
    for (int off = 128; off > 0; off >>= 1) {
        if (t < off) red[t] += red[t + off];
        __syncthreads();
    }
    const float lt = red[0];
    __syncthreads();

    // --- partial context from the LDS-resident enc tile: no 2nd HBM pass ---
    float ctx = 0.f;
#pragma unroll 4
    for (int s = 0; s < TS; ++s) ctx += sc[s] * (float)encs[s * LDE + t];
    cpart[((size_t)b * 16 + tile) * H_ + t] = ctx;
    if (t == 0) { mv[b * 16 + tile] = mt; lv[b * 16 + tile] = lt; }
}

// ---------------------------------------------------------------------------
// Kernel 3: log-sum-exp combine across the 16 s-tiles ->
//           x[b,h] = context[b,h] + y_prev[b,0,h]
// ---------------------------------------------------------------------------
__global__ __launch_bounds__(256) void combine_kernel(
    const float* __restrict__ cp, const float* __restrict__ mv,
    const float* __restrict__ lv, const float* __restrict__ y_prev,
    float* __restrict__ xb)
{
    const int b = blockIdx.x, t = threadIdx.x;
    float M = -3.402823466e38f;
#pragma unroll
    for (int i = 0; i < 16; ++i) M = fmaxf(M, mv[b * 16 + i]);
    float denom = 0.f, ctx = 0.f;
#pragma unroll
    for (int i = 0; i < 16; ++i) {
        const float w = __expf(mv[b * 16 + i] - M);
        denom += lv[b * 16 + i] * w;
        ctx   += cp[((size_t)b * 16 + i) * H_ + t] * w;
    }
    xb[(size_t)b * H_ + t] = ctx / denom + y_prev[(size_t)b * H_ + t];
}

// ---------------------------------------------------------------------------
// Kernel 4: GRU cell (PyTorch r,z,n layout) + FC head.
// out layout: d_out[0..255] = out[B,1]; d_out[256..] = h_new[1,B,H].
// ---------------------------------------------------------------------------
__global__ __launch_bounds__(256) void gru_kernel(
    const float* __restrict__ xb,  const float* __restrict__ hid,
    const float* __restrict__ W_ih, const float* __restrict__ W_hh,
    const float* __restrict__ b_ih, const float* __restrict__ b_hh,
    const float* __restrict__ Wfc,  const float* __restrict__ bfc,
    float* __restrict__ out)
{
    __shared__ float xs[H_], hs[H_], red[256];
    const int b = blockIdx.x, t = threadIdx.x;
    xs[t] = xb[(size_t)b * H_ + t];
    hs[t] = hid[(size_t)b * H_ + t];
    __syncthreads();

    float gi[3], gh[3];
#pragma unroll
    for (int g = 0; g < 3; ++g) {
        const float* wi = W_ih + (size_t)(g * H_ + t) * H_;
        const float* wh = W_hh + (size_t)(g * H_ + t) * H_;
        float si = 0.f, sh2 = 0.f;
#pragma unroll 8
        for (int e = 0; e < H_; ++e) { si += wi[e] * xs[e]; sh2 += wh[e] * hs[e]; }
        gi[g] = si  + b_ih[g * H_ + t];
        gh[g] = sh2 + b_hh[g * H_ + t];
    }
    const float r  = 1.f / (1.f + __expf(-(gi[0] + gh[0])));
    const float z  = 1.f / (1.f + __expf(-(gi[1] + gh[1])));
    const float n  = fast_tanh(gi[2] + r * gh[2]);
    const float hn = (1.f - z) * n + z * hs[t];

    out[B_ + (size_t)b * H_ + t] = hn;         // h_new

    red[t] = hn * Wfc[t];
    __syncthreads();
    for (int off = 128; off > 0; off >>= 1) {
        if (t < off) red[t] += red[t + off];
        __syncthreads();
    }
    if (t == 0) out[b] = red[0] + bfc[0];      // fc head
}

// ---------------------------------------------------------------------------
extern "C" void kernel_launch(void* const* d_in, const int* in_sizes, int n_in,
                              void* d_out, int out_size, void* d_ws, size_t ws_size,
                              hipStream_t stream)
{
    const float* y_prev = (const float*)d_in[0];
    const float* hidden = (const float*)d_in[1];   // [1,B,H] == [B,H]
    const float* enc    = (const float*)d_in[2];
    const float* Wq     = (const float*)d_in[3];
    const float* Wk     = (const float*)d_in[4];
    const float* attn_b = (const float*)d_in[5];
    const float* Ws     = (const float*)d_in[6];
    // d_in[7] = bs: additive constant over S -> softmax-invariant, unused.
    const float* W_ih   = (const float*)d_in[8];
    const float* W_hh   = (const float*)d_in[9];
    const float* b_ih   = (const float*)d_in[10];
    const float* b_hh   = (const float*)d_in[11];
    const float* Wfc    = (const float*)d_in[12];
    const float* bfc    = (const float*)d_in[13];
    float* out = (float*)d_out;

    // workspace partition (floats)
    float* ws = (float*)d_ws;
    float* qb = ws;                         // B*H            = 65536
    float* cp = qb + B_ * H_;               // B*16*H         = 1048576
    float* mv = cp + (size_t)B_ * 16 * H_;  // B*16           = 4096
    float* lv = mv + B_ * 16;               // B*16           = 4096
    float* xb = lv + B_ * 16;               // B*H            = 65536

    qb_kernel<<<B_, 256, 0, stream>>>(hidden, Wq, attn_b, qb);

    const size_t shmem = (size_t)(TS + H_) * LDE * sizeof(__bf16)
                       + (size_t)(TS + 256) * sizeof(float);   // ~204 KB of the 320 KB WGP LDS
    dim3 grid(S_ / TS, B_);
    score_ctx_kernel<<<grid, 256, shmem, stream>>>(enc, Wk, qb, Ws, cp, mv, lv);

    combine_kernel<<<B_, 256, 0, stream>>>(cp, mv, lv, y_prev, xb);

    gru_kernel<<<B_, 256, 0, stream>>>(xb, hidden, W_ih, W_hh, b_ih, b_hh,
                                       Wfc, bfc, out);
}
